// AdaptiveKeypointSelection_64269890617680
// MI455X (gfx1250) — compile-verified, matrix operand order checked by reading, not run on hardware
//
#include <hip/hip_runtime.h>

typedef __attribute__((ext_vector_type(16))) _Float16 v16h;
typedef __attribute__((ext_vector_type(8)))  _Float16 v8h;
typedef __attribute__((ext_vector_type(8)))  float    v8f;
typedef __attribute__((ext_vector_type(4)))  unsigned int u32x4;
typedef __attribute__((ext_vector_type(4)))  int      i32x4;
typedef __attribute__((ext_vector_type(8)))  int      i32x8;

#define HH 256
#define WW 256
#define HWSZ 65536

#if defined(__has_builtin)
#if __has_builtin(__builtin_amdgcn_tensor_load_to_lds) && \
    __has_builtin(__builtin_amdgcn_s_wait_tensorcnt)
#define AKP_HAVE_TDM 1
#endif
#endif

// ---------------------------------------------------------------------------
// Prep: transpose/convert weights to WMMA A-matrix order, fold BN into
// per-channel scale/shift.  A-order: k = (kh*3+kw)*Cin + ci  (ci fastest so a
// 16-wide K chunk is 32 contiguous bytes of f16).
// ---------------------------------------------------------------------------
__global__ __launch_bounds__(256) void akp_prep(
    const float* __restrict__ w1, const float* __restrict__ b1,
    const float* __restrict__ g1, const float* __restrict__ be1,
    const float* __restrict__ m1, const float* __restrict__ v1,
    const float* __restrict__ w2, const float* __restrict__ b2,
    const float* __restrict__ g2, const float* __restrict__ be2,
    const float* __restrict__ m2, const float* __restrict__ v2,
    _Float16* __restrict__ wA1, _Float16* __restrict__ wA2,
    float* __restrict__ sc1, float* __restrict__ sh1,
    float* __restrict__ sc2, float* __restrict__ sh2)
{
  int gid = blockIdx.x * 256 + threadIdx.x;
  if (gid < 64 * 1152) {               // conv1 weights: [64][9*128]
    int m = gid / 1152, k = gid % 1152;
    int khkw = k >> 7, ci = k & 127;
    int kh = khkw / 3, kw = khkw - 3 * kh;
    wA1[gid] = (_Float16)w1[((m * 128 + ci) * 3 + kh) * 3 + kw];
  }
  if (gid < 32 * 576) {                // conv2 weights: [32][9*64]
    int m = gid / 576, k = gid % 576;
    int khkw = k >> 6, ci = k & 63;
    int kh = khkw / 3, kw = khkw - 3 * kh;
    wA2[gid] = (_Float16)w2[((m * 64 + ci) * 3 + kh) * 3 + kw];
  }
  if (gid < 64) {                      // BN1 fold: y = conv*s + (be + (b-m)*s)
    float s = g1[gid] * rsqrtf(v1[gid] + 1e-5f);
    sc1[gid] = s; sh1[gid] = be1[gid] + (b1[gid] - m1[gid]) * s;
  }
  if (gid < 32) {
    float s = g2[gid] * rsqrtf(v2[gid] + 1e-5f);
    sc2[gid] = s; sh2[gid] = be2[gid] + (b2[gid] - m2[gid]) * s;
  }
}

// ---------------------------------------------------------------------------
// Conv1: 3x3, 128->64, implicit GEMM via v_wmma_f32_16x16x32_f16.
// Block = one 64-pixel row segment x all 64 Cout.  8 waves, each wave owns
// mtile (of 4) and two ntiles (of 4).  K = 9*128 = 1152 in steps of 32.
// Output act1 is NHWC f16 -> one b128 store per accumulator per lane.
// ---------------------------------------------------------------------------
#define C1_PAD 144
__global__ __launch_bounds__(256) void akp_conv1(
    const float* __restrict__ feat, const _Float16* __restrict__ wA,
    const float* __restrict__ scale, const float* __restrict__ shift,
    _Float16* __restrict__ out)
{
  __shared__ __align__(32) _Float16 smem[3 * 66 * C1_PAD];
  const int tid = threadIdx.x;
  const int w0 = blockIdx.x * 64, h = blockIdx.y, b = blockIdx.z;

  for (int i = tid; i < 3 * 66 * 128; i += 256) {
    int ci = i / 198, r2 = i - ci * 198;
    int rr = r2 / 66, cc = r2 - rr * 66;
    int hr = h + rr - 1, wc = w0 + cc - 1;
    float v = 0.f;
    if (hr >= 0 && hr < HH && wc >= 0 && wc < WW)
      v = feat[((b * 128 + ci) * HH + hr) * WW + wc];
    smem[(rr * 66 + cc) * C1_PAD + ci] = (_Float16)v;
  }
  if (h + 2 < HH && tid < 128)   // gfx1250 global_prefetch_b8 for next row
    __builtin_prefetch(&feat[((b * 128 + tid) * HH + (h + 2)) * WW + w0], 0, 1);
  __syncthreads();

  const int wave = tid >> 5, lane = tid & 31;
  const int lm = lane & 15, lhalf = lane >> 4;
  const int mtile = wave & 3, ngrp = wave >> 2;
  const _Float16* wrow = wA + (mtile * 16 + lm) * 1152;
  v8f acc0 = {}, acc1 = {};
  for (int k0 = 0; k0 < 1152; k0 += 32) {
    int ka = k0 + lhalf * 16;                 // this lane's K chunk
    v16h afrag = *(const v16h*)(wrow + ka);   // A: 16 contiguous f16
    int khkw = ka >> 7, ci0 = ka & 127;
    int rr = khkw / 3, kw = khkw - 3 * rr;
    const _Float16* bp = &smem[(rr * 66 + ngrp * 32 + lm + kw) * C1_PAD + ci0];
    v16h b0 = *(const v16h*)bp;
    v16h b1 = *(const v16h*)(bp + 16 * C1_PAD);
    acc0 = __builtin_amdgcn_wmma_f32_16x16x32_f16(false, afrag, false, b0,
                                                  (short)0, acc0, false, false);
    acc1 = __builtin_amdgcn_wmma_f32_16x16x32_f16(false, afrag, false, b1,
                                                  (short)0, acc1, false, false);
  }
  const int pix0 = w0 + ngrp * 32 + lm;
  const int mbase = mtile * 16 + lhalf * 8;
  v8h o0, o1;
  for (int r = 0; r < 8; ++r) {
    float sc = scale[mbase + r], sh = shift[mbase + r];
    o0[r] = (_Float16)fmaxf(acc0[r] * sc + sh, 0.f);
    o1[r] = (_Float16)fmaxf(acc1[r] * sc + sh, 0.f);
  }
  *(v8h*)(out + ((b * HH + h) * WW + pix0) * 64 + mbase)        = o0;
  *(v8h*)(out + ((b * HH + h) * WW + pix0 + 16) * 64 + mbase)   = o1;
}

// ---------------------------------------------------------------------------
// Conv2: 3x3, 64->32, input act1 NHWC f16.  K = 576; 8 waves = 2m x 4n tiles.
// Interior blocks stage the 3x66x64 im2col slab with the Tensor Data Mover
// (2D tile, dim0 = 66*64 contiguous f16, dim1 = 3 rows; HW pad of 8 DWORDs
// every 32 DWORDs recreates the 80-half bank-spread stride).  Boundary blocks
// (and toolchains without the TDM builtin) use vectorized v16h copies.
// ---------------------------------------------------------------------------
#define C2_PAD 80
__device__ __forceinline__ void akp_c2_fill(
    _Float16* smem, const _Float16* __restrict__ act1,
    int b, int h, int w0, int tid)
{
  for (int i = tid; i < 3 * 66 * 4; i += 256) {   // 32B chunks
    int rr = i / 264, rem = i - rr * 264;
    int cc = rem >> 2, q = rem & 3;
    int hr = h + rr - 1, wc = w0 + cc - 1;
    v16h v = {};
    if (hr >= 0 && hr < HH && wc >= 0 && wc < WW)
      v = *(const v16h*)(act1 + ((b * HH + hr) * WW + wc) * 64 + q * 16);
    *(v16h*)&smem[(rr * 66 + cc) * C2_PAD + q * 16] = v;
  }
}

__global__ __launch_bounds__(256) void akp_conv2(
    const _Float16* __restrict__ act1, const _Float16* __restrict__ wA,
    const float* __restrict__ scale, const float* __restrict__ shift,
    _Float16* __restrict__ out)
{
  __shared__ __align__(32) _Float16 smem[3 * 66 * C2_PAD];
  const int tid = threadIdx.x;
  const int w0 = blockIdx.x * 64, h = blockIdx.y, b = blockIdx.z;

#if defined(AKP_HAVE_TDM) && defined(__AMDGCN__)
  const bool interior = (h >= 1) && (h + 1 < HH) && (w0 >= 1) && (w0 + 65 <= WW);
  if (interior) {
    if (tid < 32) {                      // wave 0 issues the TDM descriptor
      unsigned lds_off = (unsigned)(size_t)&smem[0];
      unsigned long long ga = (unsigned long long)(size_t)
          (act1 + ((b * HH + (h - 1)) * WW + (w0 - 1)) * 64);
      const unsigned td0 = 66 * 64;      // dim0: 66 px * 64 ch, contiguous
      const unsigned td1 = 3;            // dim1: 3 input rows
      const unsigned stride0 = WW * 64;  // row stride in elements
      u32x4 g0 = {0u, 0u, 0u, 0u};
      g0[0] = 1u;                                  // count=1, user flags 0
      g0[1] = lds_off;                             // lds_addr (bytes)
      g0[2] = (unsigned)(ga & 0xFFFFFFFFu);        // global_addr[31:0]
      g0[3] = (unsigned)((ga >> 32) & 0x01FFFFFFu) | (2u << 30);  // type=2
      i32x8 g1 = {0, 0, 0, 0, 0, 0, 0, 0};
      // data_size=2B(1), pad_enable, pad_interval=32 DW(4), pad_amount=8 DW(7)
      g1[0] = (int)((1u << 16) | (1u << 20) | (4u << 22) | (7u << 25));
      g1[1] = (int)((td0 & 0xFFFFu) << 16);                        // dim0 lo
      g1[2] = (int)(((td0 >> 16) & 0xFFFFu) | ((td1 & 0xFFFFu) << 16));
      g1[3] = (int)(((td1 >> 16) & 0xFFFFu) | ((td0 & 0xFFFFu) << 16)); // tile0
      g1[4] = (int)(td1 & 0xFFFFu);                                // tile1
      g1[5] = (int)stride0;                                        // stride0
      i32x4 g2 = {0, 0, 0, 0}, g3 = {0, 0, 0, 0};
#if __clang_major__ >= 23
      i32x8 g4 = {0, 0, 0, 0, 0, 0, 0, 0};
      __builtin_amdgcn_tensor_load_to_lds(g0, g1, g2, g3, g4, 0);
#else
      __builtin_amdgcn_tensor_load_to_lds(g0, g1, g2, g3, 0);
#endif
      __builtin_amdgcn_s_wait_tensorcnt(0);
    }
  } else {
    akp_c2_fill(smem, act1, b, h, w0, tid);
  }
#else
  akp_c2_fill(smem, act1, b, h, w0, tid);
#endif
  __syncthreads();

  const int wave = tid >> 5, lane = tid & 31;
  const int lm = lane & 15, lhalf = lane >> 4;
  const int mtile = wave & 1, ntile = wave >> 1;
  const _Float16* wrow = wA + (mtile * 16 + lm) * 576;
  v8f acc = {};
  for (int k0 = 0; k0 < 576; k0 += 32) {
    int ka = k0 + lhalf * 16;
    v16h afrag = *(const v16h*)(wrow + ka);
    int khkw = ka >> 6, ci0 = ka & 63;
    int rr = khkw / 3, kw = khkw - 3 * rr;
    const _Float16* bp = &smem[(rr * 66 + ntile * 16 + lm + kw) * C2_PAD + ci0];
    v16h bf = *(const v16h*)bp;
    acc = __builtin_amdgcn_wmma_f32_16x16x32_f16(false, afrag, false, bf,
                                                 (short)0, acc, false, false);
  }
  const int pix = w0 + ntile * 16 + lm;
  const int mbase = mtile * 16 + lhalf * 8;
  v8h o;
  for (int r = 0; r < 8; ++r)
    o[r] = (_Float16)fmaxf(acc[r] * scale[mbase + r] + shift[mbase + r], 0.f);
  *(v8h*)(out + ((b * HH + h) * WW + pix) * 32 + mbase) = o;
}

// ---------------------------------------------------------------------------
// conv1x1 32->1 + sigmoid  ->  attention map a  (act2 is NHWC: 32 contiguous
// f16 per pixel -> two v16h loads)
// ---------------------------------------------------------------------------
__global__ __launch_bounds__(256) void akp_attmap(
    const _Float16* __restrict__ act2, const float* __restrict__ w3,
    const float* __restrict__ b3, float* __restrict__ amap)
{
  int id = blockIdx.x * 256 + threadIdx.x;   // 8*65536
  const _Float16* ap = act2 + (size_t)id * 32;
  v16h c0 = *(const v16h*)ap;
  v16h c1 = *(const v16h*)(ap + 16);
  float s = b3[0];
  for (int c = 0; c < 16; ++c) s += (float)c0[c] * w3[c];
  for (int c = 0; c < 16; ++c) s += (float)c1[c] * w3[16 + c];
  amap[id] = 1.f / (1.f + expf(-s));
}

// ---------------------------------------------------------------------------
// Coherence net: conv3x3 1->16 + ReLU, conv1x1 16->1 + sigmoid; att = a*coh
// ---------------------------------------------------------------------------
__global__ __launch_bounds__(256) void akp_coh(
    const float* __restrict__ amap, const float* __restrict__ cw1,
    const float* __restrict__ cb1, const float* __restrict__ cw2,
    const float* __restrict__ cb2, float* __restrict__ araw)
{
  __shared__ float sW1[144], sB1[16], sW2[16];
  int tid = threadIdx.x;
  if (tid < 144) sW1[tid] = cw1[tid];
  if (tid < 16) { sB1[tid] = cb1[tid]; sW2[tid] = cw2[tid]; }
  __syncthreads();
  int id = blockIdx.x * 256 + tid;
  int b = id >> 16, p = id & (HWSZ - 1);
  int y = p >> 8, x = p & 255;
  float hid[16];
  for (int j = 0; j < 16; ++j) hid[j] = 0.f;
  for (int dy = -1; dy <= 1; ++dy) {
    int yy = y + dy; if (yy < 0 || yy >= HH) continue;
    for (int dx = -1; dx <= 1; ++dx) {
      int xx = x + dx; if (xx < 0 || xx >= WW) continue;
      float av = amap[b * HWSZ + yy * WW + xx];
      int t = (dy + 1) * 3 + (dx + 1);
      for (int j = 0; j < 16; ++j) hid[j] += sW1[j * 9 + t] * av;
    }
  }
  float s = cb2[0];
  for (int j = 0; j < 16; ++j) s += fmaxf(hid[j] + sB1[j], 0.f) * sW2[j];
  float coh = 1.f / (1.f + expf(-s));
  araw[id] = amap[id] * coh;
}

// ---------------------------------------------------------------------------
// Separable 11x11 NMS max-pool: row pass then column pass + equality mask.
// ---------------------------------------------------------------------------
__global__ __launch_bounds__(256) void akp_nms_h(
    const float* __restrict__ araw, float* __restrict__ hmax)
{
  int id = blockIdx.x * 256 + threadIdx.x;
  int b = id >> 16, p = id & (HWSZ - 1);
  int y = p >> 8, x = p & 255;
  float m = -__builtin_inff();
  for (int dx = -5; dx <= 5; ++dx) {
    int xx = x + dx;
    if (xx >= 0 && xx < WW) m = fmaxf(m, araw[b * HWSZ + y * WW + xx]);
  }
  hmax[id] = m;
}

__global__ __launch_bounds__(256) void akp_nms_v(
    const float* __restrict__ araw, const float* __restrict__ hmax,
    float* __restrict__ att)
{
  int id = blockIdx.x * 256 + threadIdx.x;
  int b = id >> 16, p = id & (HWSZ - 1);
  int y = p >> 8, x = p & 255;
  float m = -__builtin_inff();
  for (int dy = -5; dy <= 5; ++dy) {
    int yy = y + dy;
    if (yy >= 0 && yy < HH) m = fmaxf(m, hmax[b * HWSZ + yy * WW + x]);
  }
  float a = araw[id];
  att[id] = (m == a) ? a : 0.f;
}

// ---------------------------------------------------------------------------
// Top-1024 per batch.  Composite 64-bit key = orderable(value)<<32 | ~index
// (unique, ties -> smaller index, matching lax.top_k).  MSB radix-select
// finds the exact 1024th key, then compaction + bitonic sort in LDS.
// ---------------------------------------------------------------------------
__device__ __forceinline__ unsigned long long akp_key(float v, unsigned p) {
  unsigned u = __float_as_uint(v);
  u = (u & 0x80000000u) ? ~u : (u | 0x80000000u);
  return ((unsigned long long)u << 32) | (unsigned long long)(~p);
}

__global__ __launch_bounds__(1024) void akp_topk(
    const float* __restrict__ att, float* __restrict__ vals,
    int* __restrict__ idxo, int* __restrict__ rowso, int* __restrict__ colso)
{
  __shared__ unsigned int hist[256];
  __shared__ unsigned long long keys[1024];
  __shared__ unsigned long long s_prefix;
  __shared__ unsigned int s_rem, s_cnt;
  const int t = threadIdx.x, b = blockIdx.x;
  const float* am = att + b * HWSZ;
  if (t == 0) { s_prefix = 0ULL; s_rem = 1024u; s_cnt = 0u; }

  for (int d = 7; d >= 0; --d) {
    if (t < 256) hist[t] = 0u;
    __syncthreads();
    unsigned long long pfx = s_prefix;
    unsigned long long mhi = (d == 7) ? 0ULL : ((~0ULL) << (8 * (d + 1)));
    for (int i = t; i < HWSZ; i += 1024) {
      unsigned long long k = akp_key(am[i], (unsigned)i);
      if ((k & mhi) == (pfx & mhi))
        atomicAdd(&hist[(unsigned)((k >> (8 * d)) & 255ULL)], 1u);
    }
    __syncthreads();
    if (t == 0) {
      unsigned rem = s_rem; int bsel = 0;
      for (int bin = 255; bin >= 0; --bin) {
        unsigned c = hist[bin];
        if (c >= rem) { bsel = bin; break; }
        rem -= c;
      }
      s_prefix = s_prefix | ((unsigned long long)bsel << (8 * d));
      s_rem = rem;
    }
    __syncthreads();
  }

  unsigned long long T = s_prefix;              // 1024th largest key
  for (int i = t; i < HWSZ; i += 1024) {
    unsigned long long k = akp_key(am[i], (unsigned)i);
    if (k >= T) {
      unsigned pos = atomicAdd(&s_cnt, 1u);
      if (pos < 1024u) keys[pos] = k;
    }
  }
  __syncthreads();
  unsigned cnt = s_cnt;
  if (t >= (int)cnt) keys[t] = 0ULL;            // defensive; cnt should be 1024
  __syncthreads();

  for (int size = 2; size <= 1024; size <<= 1)
    for (int stride = size >> 1; stride > 0; stride >>= 1) {
      __syncthreads();
      int j = t ^ stride;
      if (j > t) {
        unsigned long long a = keys[t], c = keys[j];
        bool desc = ((t & size) == 0);
        if (desc ? (a < c) : (a > c)) { keys[t] = c; keys[j] = a; }
      }
    }
  __syncthreads();

  unsigned long long k = keys[t];
  unsigned vk = (unsigned)(k >> 32);
  float v = (vk & 0x80000000u) ? __uint_as_float(vk ^ 0x80000000u)
                               : __uint_as_float(~vk);
  unsigned p = ~(unsigned)(k & 0xFFFFFFFFULL);
  vals[b * 1024 + t]  = v;
  idxo[b * 1024 + t]  = (int)p;
  rowso[b * 1024 + t] = (int)(p >> 8);
  colso[b * 1024 + t] = (int)(p & 255u);
}

// ---------------------------------------------------------------------------
// Gather descriptors at keypoints and L2-normalize along C=128.
// One 128-thread block per (b, k).
// ---------------------------------------------------------------------------
__global__ __launch_bounds__(128) void akp_gather(
    const float* __restrict__ feat, const int* __restrict__ idxo,
    float* __restrict__ sel)
{
  __shared__ float red[128];
  int kk = blockIdx.x, b = blockIdx.y, c = threadIdx.x;
  int p = idxo[b * 1024 + kk];
  float v = feat[(b * 128 + c) * HWSZ + p];
  red[c] = v * v;
  __syncthreads();
  for (int s = 64; s > 0; s >>= 1) {
    if (c < s) red[c] += red[c + s];
    __syncthreads();
  }
  float norm = fmaxf(sqrtf(red[0]), 1e-12f);
  sel[(b * 128 + c) * 1024 + kk] = v / norm;
}

// ---------------------------------------------------------------------------
extern "C" void kernel_launch(void* const* d_in, const int* in_sizes, int n_in,
                              void* d_out, int out_size, void* d_ws, size_t ws_size,
                              hipStream_t stream)
{
  (void)in_sizes; (void)n_in; (void)out_size; (void)ws_size;
  const float* feat = (const float*)d_in[0];
  const float* w1  = (const float*)d_in[1];
  const float* b1  = (const float*)d_in[2];
  const float* g1  = (const float*)d_in[3];
  const float* be1 = (const float*)d_in[4];
  const float* m1  = (const float*)d_in[5];
  const float* v1  = (const float*)d_in[6];
  const float* w2  = (const float*)d_in[7];
  const float* b2  = (const float*)d_in[8];
  const float* g2  = (const float*)d_in[9];
  const float* be2 = (const float*)d_in[10];
  const float* m2  = (const float*)d_in[11];
  const float* v2  = (const float*)d_in[12];
  const float* w3  = (const float*)d_in[13];
  const float* b3  = (const float*)d_in[14];
  const float* cw1 = (const float*)d_in[15];
  const float* cb1 = (const float*)d_in[16];
  const float* cw2 = (const float*)d_in[17];
  const float* cb2 = (const float*)d_in[18];

  // d_out layout (tuple concatenated flat, element counts):
  // sel 8*128*1024 | att 8*65536 | vals 8192 | idx 8192 | rows 8192 | cols 8192
  float* outf  = (float*)d_out;
  float* sel   = outf;
  float* att   = outf + 1048576;
  float* vals  = outf + 1572864;
  int*   idxo  = (int*)(outf + 1581056);
  int*   rowso = (int*)(outf + 1589248);
  int*   colso = (int*)(outf + 1597440);

  // workspace layout (bytes)
  char* ws = (char*)d_ws;
  _Float16* wA1  = (_Float16*)(ws + 0);                       // 147456
  _Float16* wA2  = (_Float16*)(ws + 147456);                  // 36864
  float*    sc1  = (float*)(ws + 184320);                     // 256
  float*    sh1  = (float*)(ws + 184576);                     // 256
  float*    sc2  = (float*)(ws + 184832);                     // 128
  float*    sh2  = (float*)(ws + 184960);                     // 128
  _Float16* act1 = (_Float16*)(ws + 185088);                  // NHWC, 67108864
  _Float16* act2 = (_Float16*)(ws + 185088 + 67108864);       // NHWC, 33554432
  float*    amap = (float*)(ws + 185088 + 67108864 + 33554432);
  float*    araw = amap + 8 * HWSZ;
  float*    hmax = araw + 8 * HWSZ;

  akp_prep<<<288, 256, 0, stream>>>(w1, b1, g1, be1, m1, v1,
                                    w2, b2, g2, be2, m2, v2,
                                    wA1, wA2, sc1, sh1, sc2, sh2);
  akp_conv1<<<dim3(4, 256, 8), 256, 0, stream>>>(feat, wA1, sc1, sh1, act1);
  akp_conv2<<<dim3(4, 256, 8), 256, 0, stream>>>(act1, wA2, sc2, sh2, act2);
  akp_attmap<<<2048, 256, 0, stream>>>(act2, w3, b3, amap);
  akp_coh<<<2048, 256, 0, stream>>>(amap, cw1, cb1, cw2, cb2, araw);
  akp_nms_h<<<2048, 256, 0, stream>>>(araw, hmax);
  akp_nms_v<<<2048, 256, 0, stream>>>(araw, hmax, att);
  akp_topk<<<8, 1024, 0, stream>>>(att, vals, idxo, rowso, colso);
  akp_gather<<<dim3(1024, 8), 128, 0, stream>>>(feat, idxo, sel);
}
